// ConvGRU_66305705115759
// MI455X (gfx1250) — matrix operand order repaired and checked by reading, not compiled, not run on hardware
//
#include <hip/hip_runtime.h>

#define B_ 64
#define T_ 512
#define D_ 256
#define H_ 512
#define NB_ 16   // persistent recurrence blocks (16 x 256 thr = 128 waves = 128 tiles)

typedef __attribute__((ext_vector_type(16))) _Float16 v16h;
typedef __attribute__((ext_vector_type(8)))  _Float16 v8h;
typedef __attribute__((ext_vector_type(8)))  float    v8f;

// ---------------------------------------------------------------------------
// WMMA 16x16x32 f16 -> f32 (CDNA5)
// ---------------------------------------------------------------------------
__device__ __forceinline__ v8f wmma16x16x32(v16h a, v16h b, v8f c) {
  return __builtin_amdgcn_wmma_f32_16x16x32_f16(
      /*neg_a=*/false, a, /*neg_b=*/false, b,
      /*c_mod=*/(short)0, c, /*reuse_a=*/false, /*reuse_b=*/false);
}

// Load a pre-swizzled operand register image at a (compile-time) byte offset
// from a per-lane base pointer -> lowers to global_load_b128 x2 w/ immediate.
__device__ __forceinline__ v16h ld_off(const _Float16* __restrict__ p, int byteOff) {
  return *(const v16h*)((const char*)p + byteOff);
}

__device__ __forceinline__ float sigmoidf_(float x) {
  return 1.0f / (1.0f + __expf(-x));
}

// ---------------------------------------------------------------------------
// Grid-wide generation barrier (16 resident blocks, atomics in scratch)
// ---------------------------------------------------------------------------
__device__ __forceinline__ void grid_barrier(unsigned* cnt, unsigned* gen, unsigned nb) {
  __syncthreads();
  if (threadIdx.x == 0) {
    __threadfence();
    unsigned g = __hip_atomic_load(gen, __ATOMIC_RELAXED, __HIP_MEMORY_SCOPE_AGENT);
    unsigned a = __hip_atomic_fetch_add(cnt, 1u, __ATOMIC_ACQ_REL, __HIP_MEMORY_SCOPE_AGENT);
    if (a == nb - 1u) {
      __hip_atomic_store(cnt, 0u, __ATOMIC_RELAXED, __HIP_MEMORY_SCOPE_AGENT);
      __hip_atomic_fetch_add(gen, 1u, __ATOMIC_ACQ_REL, __HIP_MEMORY_SCOPE_AGENT);
    } else {
      while (__hip_atomic_load(gen, __ATOMIC_ACQUIRE, __HIP_MEMORY_SCOPE_AGENT) == g) {
        __builtin_amdgcn_s_sleep(2);
      }
    }
    __threadfence();
  }
  __syncthreads();
}

// ---------------------------------------------------------------------------
// Prep: swizzle x/W/U into WMMA-native operand layouts (f16), zero state.
// A-operand (16x32): elem e -> k = 16*(e>>3) + 8*(lane>>4) + (e&7), m = lane&15
// B-operand (32x16): elem e -> k = 16*(e>>3) + 8*(lane>>4) + (e&7),               n = lane&15
// ---------------------------------------------------------------------------
__global__ void gru_prep(const float* __restrict__ x,
                         const float* __restrict__ Wz, const float* __restrict__ Wr,
                         const float* __restrict__ Wh, const float* __restrict__ Uz,
                         const float* __restrict__ Ur, const float* __restrict__ Uh,
                         _Float16* __restrict__ xA, _Float16* __restrict__ Wsw,
                         _Float16* __restrict__ Usw,
                         _Float16* __restrict__ hA, _Float16* __restrict__ rhA,
                         unsigned* __restrict__ bar) {
  int i = blockIdx.x * blockDim.x + threadIdx.x;
  int stride = gridDim.x * blockDim.x;

  // x -> A tiles, tIdx = mt*8 + kt   (2048 M-tiles x 8 K-tiles)
  const int XN = 2048 * 8 * 512;
  for (int j = i; j < XN; j += stride) {
    int e = j & 15, lane = (j >> 4) & 31, tIdx = j >> 9;
    int kt = tIdx & 7, mt = tIdx >> 3;
    int m = (mt << 4) + (lane & 15);
    int k = (kt << 5) + ((e >> 3) << 4) + ((lane >> 4) << 3) + (e & 7);
    xA[j] = (_Float16)x[(size_t)m * D_ + k];
  }
  // W -> B tiles, tIdx = (g*8 + kt)*32 + nt
  const int WN = 3 * 8 * 32 * 512;
  for (int j = i; j < WN; j += stride) {
    int e = j & 15, lane = (j >> 4) & 31, tIdx = j >> 9;
    int nt = tIdx & 31, gk = tIdx >> 5, kt = gk & 7, g = gk >> 3;
    int n = (nt << 4) + (lane & 15);
    int k = (kt << 5) + ((e >> 3) << 4) + ((lane >> 4) << 3) + (e & 7);
    const float* s = (g == 0) ? Wz : (g == 1 ? Wr : Wh);
    Wsw[j] = (_Float16)s[(size_t)k * H_ + n];
  }
  // U -> B tiles, tIdx = (g*16 + kt)*32 + nt
  const int UN = 3 * 16 * 32 * 512;
  for (int j = i; j < UN; j += stride) {
    int e = j & 15, lane = (j >> 4) & 31, tIdx = j >> 9;
    int nt = tIdx & 31, gk = tIdx >> 5, kt = gk & 15, g = gk >> 4;
    int n = (nt << 4) + (lane & 15);
    int k = (kt << 5) + ((e >> 3) << 4) + ((lane >> 4) << 3) + (e & 7);
    const float* s = (g == 0) ? Uz : (g == 1 ? Ur : Uh);
    Usw[j] = (_Float16)s[(size_t)k * H_ + n];
  }
  for (int j = i; j < B_ * H_; j += stride) {
    hA[j]  = (_Float16)0.0f;
    rhA[j] = (_Float16)0.0f;
  }
  if (i < 2) bar[i] = 0u;
}

// ---------------------------------------------------------------------------
// Input projections: [B*T, D] x [D, H], 3 gates, one 16x16 tile per wave.
// ---------------------------------------------------------------------------
__global__ void __launch_bounds__(256, 1)
gru_input_proj(const _Float16* __restrict__ xA, const _Float16* __restrict__ Wsw,
               _Float16* __restrict__ xzsw, _Float16* __restrict__ xrsw,
               float* __restrict__ yxh) {
  const int lane = threadIdx.x & 31;
  const int tile = (int)(((blockIdx.x << 8) + threadIdx.x) >> 5);
  const int mt = tile / 96;        // block's 8 waves share mt -> A tiles hit L0
  const int r  = tile - mt * 96;
  const int g  = r >> 5;           // gate 0..2 (wave-uniform)
  const int nt = r & 31;

  // Per-lane base pointers; kt advances by an immediate byte offset.
  const _Float16* pA = xA  + ((size_t)mt * 8) * 512 + lane * 16;        // +kt*1024B
  const _Float16* pB = Wsw + ((size_t)g * 256 + nt) * 512 + lane * 16;  // +kt*32768B

  v8f acc = {};
#pragma unroll
  for (int kt = 0; kt < 8; ++kt) {
    v16h a = ld_off(pA, kt * 1024);
    v16h b = ld_off(pB, kt * 32768);
    acc = wmma16x16x32(a, b, acc);
  }

  const int rowoff = (lane >> 4) << 3;
  const int col = lane & 15;

  if (g == 2) {
    // xh -> d_out [B,T,H]; per-v stride = H floats (immediate offsets)
    float* py = yxh + (size_t)((mt << 4) + rowoff) * H_ + (nt << 4) + col;
#pragma unroll
    for (int v = 0; v < 8; ++v) py[v * H_] = acc[v];
  } else {
    // xz/xr -> recurrence C-tile lane layout; rows of this tile are 16
    // consecutive t of the same batch, so per-v stride = 32768 halfs.
    int bidx = mt >> 5;                       // batch (uniform in tile)
    int tt0  = ((mt << 4) + rowoff) & (T_ - 1);
    int w    = (bidx >> 4) + (nt << 2);       // recurrence wave id
    int rr   = bidx & 15;
    int lane_dst = col + ((rr >> 3) << 4);
    int vv   = rr & 7;
    _Float16* dst = ((g == 0) ? xzsw : xrsw) +
                    ((((size_t)tt0 * 128 + w) * 32 + lane_dst) << 3) + vv;
#pragma unroll
    for (int v = 0; v < 8; ++v)
      *(_Float16*)((char*)dst + v * 65536) = (_Float16)acc[v];
  }
}

// ---------------------------------------------------------------------------
// Recurrence: persistent, 16 blocks x 8 waves; wave w owns tile
// (bt = w&3, nt = w>>2) of the [64,512] state for the whole sequence.
// h (f32) and z stay in VGPRs for all 512 steps; only the f16 A-operand
// images of h and r*h are shared through scratch (hA/rhA, tIdx = kt*4 + bt).
// ---------------------------------------------------------------------------
__global__ void __launch_bounds__(256, 1)
gru_recur(const _Float16* __restrict__ xzsw, const _Float16* __restrict__ xrsw,
          float* __restrict__ y, float* __restrict__ hT,
          const _Float16* __restrict__ Usw,
          const float* __restrict__ bz, const float* __restrict__ br,
          const float* __restrict__ bh,
          _Float16* __restrict__ hA, _Float16* __restrict__ rhA,
          unsigned* __restrict__ bar) {
  const int lane = threadIdx.x & 31;
  const int wave = (int)(((blockIdx.x << 8) + threadIdx.x) >> 5);  // 0..127
  const int bt = wave & 3;
  const int nt = wave >> 2;
  const int m0 = bt << 4;
  const int n0 = nt << 4;

  const int rowoff = (lane >> 4) << 3;
  const int col = lane & 15;
  const float bzc = bz[n0 + col];
  const float brc = br[n0 + col];
  const float bhc = bh[n0 + col];

  unsigned* cnt = bar;
  unsigned* gen = bar + 1;

  // Per-lane base pointers for the GEMM streams (kt stride is an immediate).
  const _Float16* pHA  = hA  + (size_t)bt * 512 + lane * 16;             // +kt*4096B
  const _Float16* pRHA = rhA + (size_t)bt * 512 + lane * 16;             // +kt*4096B
  const _Float16* pUz  = Usw + ((size_t)0   + nt) * 512 + lane * 16;     // +kt*32768B
  const _Float16* pUr  = Usw + ((size_t)512 + nt) * 512 + lane * 16;
  const _Float16* pUh  = Usw + ((size_t)1024 + nt) * 512 + lane * 16;

  // Scatter base into the A-operand image for this lane's 8 state elems:
  // slots are aoff0 + v*16 halfs (v*32 bytes) — immediate-offset stores.
  int kt_n = (n0 + col) >> 5, kl = (n0 + col) & 31;
  int kb = (kl >> 3) & 1;
  int e0 = (kl & 7) + ((kl >> 4) << 3);
  int aoff0 = (((kt_n << 2) + bt) * 32 + rowoff + (kb << 4)) * 16 + e0;
  _Float16* pHst  = hA  + aoff0;
  _Float16* pRHst = rhA + aoff0;

  // xz/xr gather base (v8h per lane per step; +t*65536B per step)
  const _Float16* pXZ = xzsw + (((size_t)wave * 32 + lane) << 3);
  const _Float16* pXR = xrsw + (((size_t)wave * 32 + lane) << 3);

  // y base: per-v stride = T*H*4 = 1MB (immediate), +t*2048B per step
  float* py = y + (size_t)(m0 + rowoff) * (T_ * H_) + n0 + col;

  v8f hreg = {};   // fp32 h_prev tile — register-resident for all 512 steps

  for (int t = 0; t < T_; ++t) {
    // -------- phase A: z, r gates; publish r*h (A-operand image) --------
    v8f accz = {}, accr = {};
#pragma unroll
    for (int kt = 0; kt < 16; ++kt) {
      v16h a  = ld_off(pHA, kt * 4096);
      v16h b0 = ld_off(pUz, kt * 32768);
      v16h b1 = ld_off(pUr, kt * 32768);
      accz = wmma16x16x32(a, b0, accz);
      accr = wmma16x16x32(a, b1, accr);
    }
    v8h xz8 = *(const v8h*)((const char*)pXZ + (size_t)t * 65536);
    v8h xr8 = *(const v8h*)((const char*)pXR + (size_t)t * 65536);
    v8f zreg = {};
#pragma unroll
    for (int v = 0; v < 8; ++v) {
      float zv = sigmoidf_(accz[v] + (float)xz8[v] + bzc);
      float rv = sigmoidf_(accr[v] + (float)xr8[v] + brc);
      zreg[v] = zv;
      *(_Float16*)((char*)pRHst + v * 32) = (_Float16)(rv * hreg[v]);
    }
    grid_barrier(cnt, gen, NB_);

    // -------- phase B: candidate h~, state update, output --------
    v8f acch = {};
#pragma unroll
    for (int kt = 0; kt < 16; ++kt) {
      v16h a = ld_off(pRHA, kt * 4096);
      v16h b = ld_off(pUh, kt * 32768);
      acch = wmma16x16x32(a, b, acch);
    }
    float* pyt = py + (size_t)t * H_;
#pragma unroll
    for (int v = 0; v < 8; ++v) {
      float* pe = (float*)((char*)pyt + (size_t)v * (T_ * H_ * 4));
      float xhv = *pe;                         // staged xh projection
      float htl = tanhf(acch[v] + xhv + bhc);
      float hn  = (1.0f - zreg[v]) * htl + zreg[v] * hreg[v];
      hreg[v] = hn;
      *pe = hn;                                // overwrite with output
      *(_Float16*)((char*)pHst + v * 32) = (_Float16)hn;  // next A image
    }
    if (t == T_ - 1) {
#pragma unroll
      for (int v = 0; v < 8; ++v)
        hT[(size_t)(m0 + rowoff + v) * H_ + n0 + col] = hreg[v];
    }
    grid_barrier(cnt, gen, NB_);
  }
}

// ---------------------------------------------------------------------------
extern "C" void kernel_launch(void* const* d_in, const int* in_sizes, int n_in,
                              void* d_out, int out_size, void* d_ws, size_t ws_size,
                              hipStream_t stream) {
  const float* x  = (const float*)d_in[0];
  const float* Wz = (const float*)d_in[1];
  const float* Wr = (const float*)d_in[2];
  const float* Wh = (const float*)d_in[3];
  const float* Uz = (const float*)d_in[4];
  const float* Ur = (const float*)d_in[5];
  const float* Uh = (const float*)d_in[6];
  const float* bz = (const float*)d_in[7];
  const float* br = (const float*)d_in[8];
  const float* bh = (const float*)d_in[9];

  char* ws = (char*)d_ws;
  size_t off = 0;
  unsigned*  bar  = (unsigned*)(ws + off);  off += 256;
  _Float16*  hA   = (_Float16*)(ws + off);  off += (size_t)B_ * H_ * 2;
  _Float16*  rhA  = (_Float16*)(ws + off);  off += (size_t)B_ * H_ * 2;
  _Float16*  Usw  = (_Float16*)(ws + off);  off += (size_t)3 * H_ * H_ * 2;
  _Float16*  Wsw  = (_Float16*)(ws + off);  off += (size_t)3 * D_ * H_ * 2;
  _Float16*  xA   = (_Float16*)(ws + off);  off += (size_t)B_ * T_ * D_ * 2;
  _Float16*  xzsw = (_Float16*)(ws + off);  off += (size_t)T_ * B_ * H_ * 2;
  _Float16*  xrsw = (_Float16*)(ws + off);  off += (size_t)T_ * B_ * H_ * 2;

  float* y  = (float*)d_out;
  float* hT = y + (size_t)B_ * T_ * H_;

  gru_prep<<<8192, 256, 0, stream>>>(x, Wz, Wr, Wh, Uz, Ur, Uh,
                                     xA, Wsw, Usw, hA, rhA, bar);
  // tiles: 2048 (M) * 32 (N) * 3 gates = 196608 waves -> 24576 blocks of 8 waves
  gru_input_proj<<<24576, 256, 0, stream>>>(xA, Wsw, xzsw, xrsw, y);
  gru_recur<<<NB_, 256, 0, stream>>>(xzsw, xrsw, y, hT, Usw, bz, br, bh,
                                     hA, rhA, bar);
}
